// AttetionLayer_45457933861476
// MI455X (gfx1250) — matrix operand order repaired and checked by reading, not compiled, not run on hardware
//
#include <hip/hip_runtime.h>

#define B_  4
#define C_  256
#define CI_ 32
#define N_  4096

typedef __bf16 bf16_t;
typedef bf16_t v16bf __attribute__((ext_vector_type(16)));
typedef bf16_t v8bf  __attribute__((ext_vector_type(8)));
typedef float  v8f   __attribute__((ext_vector_type(8)));

union V16U { v16bf v; v8bf h[2]; };
union PKU  { v8bf v; int i[4]; };
union FIU  { float f; int i; };

// ---------------------------------------------------------------------------
// Cross-half-wave exchange (lane ^ 16) as a single VALU op (v_permlanex16_b32)
// instead of a ds_bpermute LDS round-trip. Fallback: __shfl_xor.
// ---------------------------------------------------------------------------
#if defined(__HIP_DEVICE_COMPILE__) && __has_builtin(__builtin_amdgcn_permlanex16)
__device__ __forceinline__ int xor16_i32(int v) {
    return __builtin_amdgcn_permlanex16(v, v, 0x76543210, 0xfedcba98, false, false);
}
#else
__device__ __forceinline__ int xor16_i32(int v) { return __shfl_xor(v, 16, 32); }
#endif
__device__ __forceinline__ float xor16_f32(float v) {
    FIU u; u.f = v; u.i = xor16_i32(u.i); return u.f;
}

// ---------------------------------------------------------------------------
// Kernel 1: f/g/h projections (fp32 compute, bf16 outputs laid out for WMMA)
//   fT, gT : (B, N, 32)  bf16   (row n contiguous over channels)
//   hC     : (B, 32, N)  bf16   (row c contiguous over sequence)
// ---------------------------------------------------------------------------
__global__ __launch_bounds__(256) void proj_kernel(
    const float* __restrict__ x,
    const float* __restrict__ wf, const float* __restrict__ bf,
    const float* __restrict__ wg, const float* __restrict__ bg,
    const float* __restrict__ wh, const float* __restrict__ bh,
    bf16_t* __restrict__ fT, bf16_t* __restrict__ gT, bf16_t* __restrict__ hC)
{
    __shared__ float xs[C_ * 32];    // 32 KB: x tile (C x 32 positions)
    __shared__ float wsh[24 * C_];   // 24 KB: 24 rows of the virtual (96 x 256) weight
    const int t  = threadIdx.x;
    const int n0 = blockIdx.x * 32;
    const int b  = blockIdx.y;
    const int z  = blockIdx.z;       // channels z*24 .. z*24+23 of [f;g;h]

    for (int i = t; i < 24 * C_; i += 256) {
        int row = i >> 8, c = i & (C_ - 1);
        int o = z * 24 + row;
        int p = o >> 5, oc = o & 31;
        const float* wsrc = (p == 0) ? wf : (p == 1) ? wg : wh;
        wsh[i] = wsrc[oc * C_ + c];
    }
    const float* xb = x + (size_t)b * C_ * N_;
    for (int i = t; i < C_ * 8; i += 256) {
        int c = i >> 3, n4 = i & 7;
        *(float4*)(xs + c * 32 + n4 * 4) =
            *(const float4*)(xb + (size_t)c * N_ + n0 + n4 * 4);
    }
    __syncthreads();

    const int n  = t & 31;
    const int og = t >> 5;           // wave id -> 3 output channels each
    float acc[3] = {0.f, 0.f, 0.f};
    for (int c = 0; c < C_; ++c) {
        float xv = xs[c * 32 + n];
        acc[0] = fmaf(wsh[(og * 3 + 0) * C_ + c], xv, acc[0]);
        acc[1] = fmaf(wsh[(og * 3 + 1) * C_ + c], xv, acc[1]);
        acc[2] = fmaf(wsh[(og * 3 + 2) * C_ + c], xv, acc[2]);
    }
    const int gn = n0 + n;
#pragma unroll
    for (int j = 0; j < 3; ++j) {
        int o = z * 24 + og * 3 + j;
        int p = o >> 5, oc = o & 31;
        if (p == 0)
            fT[((size_t)b * N_ + gn) * CI_ + oc] = (bf16_t)(acc[j] + bf[oc]);
        else if (p == 1)
            gT[((size_t)b * N_ + gn) * CI_ + oc] = (bf16_t)(acc[j] + bg[oc]);
        else
            hC[((size_t)b * CI_ + oc) * N_ + gn] = (bf16_t)(acc[j] + bh[oc]);
    }
}

// ---------------------------------------------------------------------------
// Kernel 2: fused flash attention, one wave = 16 queries, register-resident.
//   S^T(key,q) = sum_c g[c,key] f[c,q]  via v_wmma_f32_16x16x32_bf16
//   online softmax over keys (within-lane + permlanex16 pair combine)
//   O^T(c,q)  += h(c,keys) * P^T(keys,q)
//   x0T: (B, N, 32) fp32
// ---------------------------------------------------------------------------
__global__ __launch_bounds__(256) void attn_kernel(
    const bf16_t* __restrict__ fT, const bf16_t* __restrict__ gT,
    const bf16_t* __restrict__ hC, float* __restrict__ x0T)
{
    const int b    = blockIdx.y;
    const int lane = threadIdx.x & 31;
    const int wave = threadIdx.x >> 5;                 // 0..7
    const int q0   = blockIdx.x * 128 + wave * 16;
    const int ql   = lane & 15;
    const bool lo  = lane < 16;

    const bf16_t* fTb = fT + (size_t)b * N_ * CI_;
    const bf16_t* gTb = gT + (size_t)b * N_ * CI_;
    const bf16_t* hb  = hC + (size_t)b * CI_ * N_;

    // B operand (f): lane holds channels c0..c0+7 and c0+16..c0+23 of its query column
    V16U fq;
    {
        const int c0 = lo ? 0 : 8;
        fq.h[0] = *(const v8bf*)(fTb + (size_t)(q0 + ql) * CI_ + c0);
        fq.h[1] = *(const v8bf*)(fTb + (size_t)(q0 + ql) * CI_ + c0 + 16);
    }

    v8f o0 = {0, 0, 0, 0, 0, 0, 0, 0};   // O^T rows c=0..15
    v8f o1 = {0, 0, 0, 0, 0, 0, 0, 0};   // O^T rows c=16..31
    float m = -__builtin_inff();
    float lsum = 0.f;

    for (int kt = 0; kt < N_; kt += 32) {
        if (kt + 32 < N_) {
            __builtin_prefetch(gTb + (size_t)(kt + 32 + ql) * CI_, 0, 1);
            __builtin_prefetch(hb + (size_t)ql * N_ + kt + 32, 0, 1);
        }
        // A operands from g^T: rows = keys, K = channels
        V16U ag0, ag1;
        {
            const int cb = lo ? 0 : 8;
            const bf16_t* p = gTb + (size_t)(kt + ql) * CI_ + cb;
            ag0.h[0] = *(const v8bf*)(p);
            ag0.h[1] = *(const v8bf*)(p + 16);
            p += 16 * CI_;
            ag1.h[0] = *(const v8bf*)(p);
            ag1.h[1] = *(const v8bf*)(p + 16);
        }
        // A operands from h: rows = channels, K = keys (independent of softmax,
        // issued early so the loads overlap the S-WMMAs + softmax chain)
        V16U ah0, ah1;
        {
            const int ko = lo ? 0 : 8;
            const bf16_t* p = hb + (size_t)ql * N_ + kt + ko;
            ah0.h[0] = *(const v8bf*)(p);
            ah0.h[1] = *(const v8bf*)(p + 16);
            p = hb + (size_t)(16 + ql) * N_ + kt + ko;
            ah1.h[0] = *(const v8bf*)(p);
            ah1.h[1] = *(const v8bf*)(p + 16);
        }

        v8f z = {0, 0, 0, 0, 0, 0, 0, 0};
        v8f st0 = __builtin_amdgcn_wmma_f32_16x16x32_bf16(
            false, ag0.v, false, fq.v, (short)0, z, false, false);
        v8f st1 = __builtin_amdgcn_wmma_f32_16x16x32_bf16(
            false, ag1.v, false, fq.v, (short)0, z, false, false);

        // ---- online softmax (this lane holds 16 key-logits of its query) ----
        float tmax = st0[0];
#pragma unroll
        for (int r = 1; r < 8; ++r) tmax = fmaxf(tmax, st0[r]);
#pragma unroll
        for (int r = 0; r < 8; ++r) tmax = fmaxf(tmax, st1[r]);
        tmax = fmaxf(tmax, xor16_f32(tmax));
        const float mnew  = fmaxf(m, tmax);
        const float scale = __expf(m - mnew);
        float p0[8], p1[8], tsum = 0.f;
#pragma unroll
        for (int r = 0; r < 8; ++r) { p0[r] = __expf(st0[r] - mnew); tsum += p0[r]; }
#pragma unroll
        for (int r = 0; r < 8; ++r) { p1[r] = __expf(st1[r] - mnew); tsum += p1[r]; }
        tsum += xor16_f32(tsum);
        lsum = lsum * scale + tsum;
        m = mnew;
#pragma unroll
        for (int r = 0; r < 8; ++r) { o0[r] *= scale; o1[r] *= scale; }

        // ---- build P^T B-operand (keys x queries) directly from p0/p1 (no swap needed) ----
        PKU pb0, pb1, sw0, sw1;
#pragma unroll
        for (int r = 0; r < 8; ++r) {
            pb0.v[r] = (bf16_t)p0[r];
            pb1.v[r] = (bf16_t)p1[r];
        }
#pragma unroll
        for (int k = 0; k < 4; ++k) {
            sw0.i[k] = xor16_i32(pb0.i[k]);
            sw1.i[k] = xor16_i32(pb1.i[k]);
        }
        V16U bp;
        bp.h[0] = pb0.v;   // keys kt+0..7   | kt+8..15
        bp.h[1] = pb1.v;   // keys kt+16..23 | kt+24..31

        o0 = __builtin_amdgcn_wmma_f32_16x16x32_bf16(
            false, ah0.v, false, bp.v, (short)0, o0, false, false);
        o1 = __builtin_amdgcn_wmma_f32_16x16x32_bf16(
            false, ah1.v, false, bp.v, (short)0, o1, false, false);
    }

    const float inv = 1.0f / lsum;
#pragma unroll
    for (int r = 0; r < 8; ++r) { o0[r] *= inv; o1[r] *= inv; }

    float* outp = x0T + ((size_t)b * N_ + q0 + ql) * CI_ + (lo ? 0 : 8);
    *(v8f*)(outp)      = o0;
    *(v8f*)(outp + 16) = o1;
}

// ---------------------------------------------------------------------------
// Kernel 3: v = wv @ x0 + bv ; out = gamma*v + x
// ---------------------------------------------------------------------------
__global__ __launch_bounds__(256) void out_kernel(
    const float* __restrict__ x0T, const float* __restrict__ wv,
    const float* __restrict__ bv, const float* __restrict__ gamma,
    const float* __restrict__ x, float* __restrict__ out)
{
    __shared__ float wvs[C_ * CI_];  // 32 KB
    __shared__ float bvs[C_];
    const int t  = threadIdx.x;
    const int n0 = blockIdx.x * 64;
    const int b  = blockIdx.y;

    for (int i = t; i < C_ * CI_ / 4; i += 256)
        ((float4*)wvs)[i] = ((const float4*)wv)[i];
    if (t < C_) bvs[t] = bv[t];
    __syncthreads();

    const float g  = gamma[0];
    const int   n  = t & 63;
    const int   cg = t >> 6;         // 0..3 -> 64 channels each

    float xv[CI_];
    const float4* xr = (const float4*)(x0T + ((size_t)b * N_ + n0 + n) * CI_);
#pragma unroll
    for (int i = 0; i < CI_ / 4; ++i) {
        float4 v4 = xr[i];
        xv[4 * i + 0] = v4.x; xv[4 * i + 1] = v4.y;
        xv[4 * i + 2] = v4.z; xv[4 * i + 3] = v4.w;
    }
    for (int c = cg * 64; c < cg * 64 + 64; ++c) {
        float acc = bvs[c];
        const float* wr = wvs + c * CI_;
#pragma unroll
        for (int i = 0; i < CI_; ++i) acc = fmaf(wr[i], xv[i], acc);
        size_t idx = ((size_t)b * C_ + c) * N_ + n0 + n;
        out[idx] = g * acc + x[idx];
    }
}

// ---------------------------------------------------------------------------
extern "C" void kernel_launch(void* const* d_in, const int* in_sizes, int n_in,
                              void* d_out, int out_size, void* d_ws, size_t ws_size,
                              hipStream_t stream)
{
    const float* x     = (const float*)d_in[0];
    const float* wf    = (const float*)d_in[1];
    const float* bf    = (const float*)d_in[2];
    const float* wg    = (const float*)d_in[3];
    const float* bg    = (const float*)d_in[4];
    const float* wh    = (const float*)d_in[5];
    const float* bh    = (const float*)d_in[6];
    const float* wv    = (const float*)d_in[7];
    const float* bv    = (const float*)d_in[8];
    const float* gamma = (const float*)d_in[9];

    char* ws = (char*)d_ws;
    bf16_t* fT  = (bf16_t*)(ws);                 // 1 MB
    bf16_t* gT  = (bf16_t*)(ws + (1u << 20));    // 1 MB
    bf16_t* hC  = (bf16_t*)(ws + (2u << 20));    // 1 MB
    float*  x0T = (float*)(ws + (3u << 20));     // 2 MB

    proj_kernel<<<dim3(N_ / 32, B_, 4), 256, 0, stream>>>(
        x, wf, bf, wg, bg, wh, bh, fT, gT, hC);
    attn_kernel<<<dim3(N_ / 128, B_), 256, 0, stream>>>(fT, gT, hC, x0T);
    out_kernel<<<dim3(N_ / 64, B_), 256, 0, stream>>>(
        x0T, wv, bv, gamma, x, (float*)d_out);
}
